// SelectiveSSM_29807073034788
// MI455X (gfx1250) — compile-verified, matrix operand order checked by reading, not run on hardware
//
#include <hip/hip_runtime.h>

// ---------------- Types for CDNA5 WMMA ----------------
typedef __bf16 bf16_t;
typedef __attribute__((ext_vector_type(16))) __bf16 v16bf;
typedef __attribute__((ext_vector_type(8)))  float  v8f;

// Problem constants (from reference)
#define BATCH   2
#define SEQL    2048
#define DMODEL  1024
#define DINNER  2048
#define DSTATE  16
#define MROWS   (BATCH * SEQL)   // 4096

// ---------------- f32 -> bf16 convert ----------------
__global__ void cvt_bf16_kernel(const float* __restrict__ in, bf16_t* __restrict__ out, int n) {
    int t = blockIdx.x * blockDim.x + threadIdx.x;
    if (t < n) out[t] = (bf16_t)in[t];
}

// ---------------- WMMA fragment load ----------------
// Loads a 16x32 bf16 tile (row-major, leading dim = ldk) into the CDNA5
// 16-bit A/B fragment layout: lane l (l<16) holds row l, K=[8h,8h+8) in
// v[0..3] and K=[16+8h,16+8h+8) in v[4..7], h = lane>>4. Two b128 loads.
__device__ __forceinline__ v16bf load_frag_bf16(const bf16_t* __restrict__ p,
                                                int row0, int ldk, int k0, int lane) {
    int half = (lane >> 4) & 1;
    int r    = lane & 15;
    const bf16_t* base = p + (size_t)(row0 + r) * ldk + k0 + half * 8;
    v16bf f;
    reinterpret_cast<uint4*>(&f)[0] = *reinterpret_cast<const uint4*>(base);
    reinterpret_cast<uint4*>(&f)[1] = *reinterpret_cast<const uint4*>(base + 16);
    return f;
}

// ---------------- C/D tile store (+optional softplus epilogue) ----------------
// 32-bit C/D layout: VGPR r, lanes 0-15 -> (M=r, N=lane); lanes 16-31 -> (M=r+8).
template <int EPI>
__device__ __forceinline__ void store_tile(float* __restrict__ C, int ldn,
                                           int m0, int n0, v8f acc, int lane,
                                           const float* __restrict__ bias) {
    int n  = lane & 15;
    int mh = (lane >> 4) * 8;
    float bv = 0.0f;
    if (EPI == 1) bv = bias[n0 + n];
#pragma unroll
    for (int r = 0; r < 8; ++r) {
        float v = acc[r];
        if (EPI == 1) {           // softplus(acc + bias)
            v += bv;
            v = (v > 20.0f) ? v : __logf(1.0f + __expf(v));
        }
        C[(size_t)(m0 + mh + r) * ldn + n0 + n] = v;
    }
}

// ---------------- Generic GEMM:  C[M,N] = A_bf16[M,K] @ W_bf16[N,K]^T ----------------
// Block = 128 threads = 4 waves stacked along M. Wave tile = 64x32 (4x2 WMMA
// tiles, 8 f32 accumulators): 8 v_wmma per 12 global_load_b128 (1.5 loads/wmma).
template <int EPI>
__global__ void wmma_gemm_kernel(const bf16_t* __restrict__ A,
                                 const bf16_t* __restrict__ W,
                                 float* __restrict__ C,
                                 const float* __restrict__ bias,
                                 int M, int N, int K) {
    int lane = threadIdx.x & 31;
    int wave = threadIdx.x >> 5;
    int m0 = blockIdx.y * 256 + wave * 64;
    int n0 = blockIdx.x * 32;

    v8f c00 = {}; v8f c01 = {};
    v8f c10 = {}; v8f c11 = {};
    v8f c20 = {}; v8f c21 = {};
    v8f c30 = {}; v8f c31 = {};

    for (int k0 = 0; k0 < K; k0 += 32) {
        v16bf a0 = load_frag_bf16(A, m0,      K, k0, lane);
        v16bf a1 = load_frag_bf16(A, m0 + 16, K, k0, lane);
        v16bf a2 = load_frag_bf16(A, m0 + 32, K, k0, lane);
        v16bf a3 = load_frag_bf16(A, m0 + 48, K, k0, lane);
        v16bf b0 = load_frag_bf16(W, n0,      K, k0, lane);
        v16bf b1 = load_frag_bf16(W, n0 + 16, K, k0, lane);
        if (k0 + 32 < K) {  // uniform branch, EXEC stays all-1s; WGP-scope prefetch
            __builtin_prefetch(A + (size_t)(m0 + 16 + (lane & 15)) * K + k0 + 32, 0, 3);
            __builtin_prefetch(W + (size_t)(n0 + (lane & 15)) * K + k0 + 32, 0, 3);
        }
        c00 = __builtin_amdgcn_wmma_f32_16x16x32_bf16(false, a0, false, b0, (short)0, c00, false, false);
        c01 = __builtin_amdgcn_wmma_f32_16x16x32_bf16(false, a0, false, b1, (short)0, c01, false, false);
        c10 = __builtin_amdgcn_wmma_f32_16x16x32_bf16(false, a1, false, b0, (short)0, c10, false, false);
        c11 = __builtin_amdgcn_wmma_f32_16x16x32_bf16(false, a1, false, b1, (short)0, c11, false, false);
        c20 = __builtin_amdgcn_wmma_f32_16x16x32_bf16(false, a2, false, b0, (short)0, c20, false, false);
        c21 = __builtin_amdgcn_wmma_f32_16x16x32_bf16(false, a2, false, b1, (short)0, c21, false, false);
        c30 = __builtin_amdgcn_wmma_f32_16x16x32_bf16(false, a3, false, b0, (short)0, c30, false, false);
        c31 = __builtin_amdgcn_wmma_f32_16x16x32_bf16(false, a3, false, b1, (short)0, c31, false, false);
    }

    store_tile<EPI>(C, N, m0,      n0,      c00, lane, bias);
    store_tile<EPI>(C, N, m0,      n0 + 16, c01, lane, bias);
    store_tile<EPI>(C, N, m0 + 16, n0,      c10, lane, bias);
    store_tile<EPI>(C, N, m0 + 16, n0 + 16, c11, lane, bias);
    store_tile<EPI>(C, N, m0 + 32, n0,      c20, lane, bias);
    store_tile<EPI>(C, N, m0 + 32, n0 + 16, c21, lane, bias);
    store_tile<EPI>(C, N, m0 + 48, n0,      c30, lane, bias);
    store_tile<EPI>(C, N, m0 + 48, n0 + 16, c31, lane, bias);
}

// ---------------- depthwise causal conv1d(k=4) + SiLU ----------------
// xz is (B*L, 4096): x_branch = [:,0:2048]. Writes xc fp32 and xc bf16.
__global__ void conv_silu_kernel(const float* __restrict__ xz,
                                 const float* __restrict__ cw,
                                 const float* __restrict__ cb,
                                 float* __restrict__ xc,
                                 bf16_t* __restrict__ xc_bf) {
    int t = blockIdx.x * blockDim.x + threadIdx.x;   // [0, B*L*DINNER)
    int d  = t % DINNER;
    int bl = t / DINNER;
    int l  = bl % SEQL;
    int b  = bl / SEQL;
    float acc = cb[d];
#pragma unroll
    for (int j = 0; j < 4; ++j) {
        int ls = l + j - 3;
        if (ls >= 0)
            acc += xz[(size_t)(b * SEQL + ls) * (2 * DINNER) + d] * cw[d * 4 + j];
    }
    float v = acc / (1.0f + __expf(-acc));           // silu
    xc[t]    = v;
    xc_bf[t] = (bf16_t)v;
}

// ---------------- Selective scan (sequential in L) ----------------
// One lane per (b, d, s).  16-lane shuffle reduction over states; lane s==0
// fuses the gate y*silu(z) + xc*D and writes bf16 for the final out_proj GEMM.
__global__ void scan_kernel(const float* __restrict__ xc,
                            const float* __restrict__ dt,
                            const float* __restrict__ xdbl,   // (B*L, 32): B | C
                            const float* __restrict__ A_log,
                            const float* __restrict__ Dvec,
                            const float* __restrict__ xz,     // z = xz[:, 2048+d]
                            bf16_t* __restrict__ y_bf) {
    int t = blockIdx.x * blockDim.x + threadIdx.x;   // [0, B*DINNER*16)
    int s = t & 15;
    int g = t >> 4;
    int d = g % DINNER;
    int b = g / DINNER;

    float A  = -__expf(A_log[d * DSTATE + s]);
    float Dd = Dvec[d];
    float h  = 0.0f;

    for (int l = 0; l < SEQL; ++l) {
        int bl = b * SEQL + l;
        float dtv = dt[(size_t)bl * DINNER + d];
        float xv  = xc[(size_t)bl * DINNER + d];
        float Bv  = xdbl[bl * 32 + s];
        float Cv  = xdbl[bl * 32 + 16 + s];
        h = __expf(A * dtv) * h + Bv * (xv * dtv);
        float py = h * Cv;
        py += __shfl_xor(py, 8, 16);
        py += __shfl_xor(py, 4, 16);
        py += __shfl_xor(py, 2, 16);
        py += __shfl_xor(py, 1, 16);
        if (s == 0) {
            float z  = xz[(size_t)bl * (2 * DINNER) + DINNER + d];
            float yo = py * (z / (1.0f + __expf(-z))) + xv * Dd;
            y_bf[(size_t)bl * DINNER + d] = (bf16_t)yo;
        }
    }
}

// ---------------- Host launch ----------------
extern "C" void kernel_launch(void* const* d_in, const int* in_sizes, int n_in,
                              void* d_out, int out_size, void* d_ws, size_t ws_size,
                              hipStream_t stream) {
    const float* x         = (const float*)d_in[0];
    const float* in_proj_w = (const float*)d_in[1];
    const float* conv_w    = (const float*)d_in[2];
    const float* conv_b    = (const float*)d_in[3];
    const float* x_proj_w  = (const float*)d_in[4];
    const float* dt_proj_w = (const float*)d_in[5];
    const float* dt_proj_b = (const float*)d_in[6];
    const float* A_log     = (const float*)d_in[7];
    const float* Dvec      = (const float*)d_in[8];
    const float* out_proj_w= (const float*)d_in[9];

    // Workspace carve-out (256B aligned)
    char* p = (char*)d_ws;
    auto alloc = [&p](size_t bytes) -> void* {
        void* r = (void*)p;
        p += (bytes + 255) & ~(size_t)255;
        return r;
    };
    bf16_t* x_bf     = (bf16_t*)alloc((size_t)MROWS * DMODEL * 2);
    bf16_t* w_in_bf  = (bf16_t*)alloc((size_t)2 * DINNER * DMODEL * 2);
    bf16_t* w_xp_bf  = (bf16_t*)alloc((size_t)2 * DSTATE * DINNER * 2);
    bf16_t* w_dt_bf  = (bf16_t*)alloc((size_t)DINNER * DINNER * 2);
    bf16_t* w_out_bf = (bf16_t*)alloc((size_t)DMODEL * DINNER * 2);
    float*  xz       = (float*) alloc((size_t)MROWS * 2 * DINNER * 4);
    float*  xc       = (float*) alloc((size_t)MROWS * DINNER * 4);
    bf16_t* xc_bf    = (bf16_t*)alloc((size_t)MROWS * DINNER * 2);
    float*  xdbl     = (float*) alloc((size_t)MROWS * 2 * DSTATE * 4);
    float*  dt       = (float*) alloc((size_t)MROWS * DINNER * 4);
    bf16_t* y_bf     = (bf16_t*)alloc((size_t)MROWS * DINNER * 2);

    // 0) bf16 conversions of GEMM operands
    auto cvt = [&](const float* src, bf16_t* dst, int n) {
        cvt_bf16_kernel<<<(n + 255) / 256, 256, 0, stream>>>(src, dst, n);
    };
    cvt(x,          x_bf,     MROWS * DMODEL);
    cvt(in_proj_w,  w_in_bf,  2 * DINNER * DMODEL);
    cvt(x_proj_w,   w_xp_bf,  2 * DSTATE * DINNER);
    cvt(dt_proj_w,  w_dt_bf,  DINNER * DINNER);
    cvt(out_proj_w, w_out_bf, DMODEL * DINNER);

    dim3 blk(128);

    // 1) in_proj: xz = x @ in_proj_w^T   (M=4096, N=4096, K=1024)
    wmma_gemm_kernel<0><<<dim3((2 * DINNER) / 32, MROWS / 256), blk, 0, stream>>>(
        x_bf, w_in_bf, xz, nullptr, MROWS, 2 * DINNER, DMODEL);

    // 2) depthwise causal conv + silu
    conv_silu_kernel<<<(MROWS * DINNER) / 256, 256, 0, stream>>>(
        xz, conv_w, conv_b, xc, xc_bf);

    // 3) x_proj: x_dbl = xc @ x_proj_w^T  (N=32)
    wmma_gemm_kernel<0><<<dim3((2 * DSTATE) / 32, MROWS / 256), blk, 0, stream>>>(
        xc_bf, w_xp_bf, xdbl, nullptr, MROWS, 2 * DSTATE, DINNER);

    // 4) dt_proj + softplus epilogue  (N=2048, K=2048)
    wmma_gemm_kernel<1><<<dim3(DINNER / 32, MROWS / 256), blk, 0, stream>>>(
        xc_bf, w_dt_bf, dt, dt_proj_b, MROWS, DINNER, DINNER);

    // 5) selective scan + fused gate, writes bf16 y
    scan_kernel<<<(BATCH * DINNER * DSTATE) / 256, 256, 0, stream>>>(
        xc, dt, xdbl, A_log, Dvec, xz, y_bf);

    // 6) out_proj: out = y @ out_proj_w^T  (N=1024, K=2048) -> fp32 d_out
    wmma_gemm_kernel<0><<<dim3(DMODEL / 32, MROWS / 256), blk, 0, stream>>>(
        y_bf, w_out_bf, (float*)d_out, nullptr, MROWS, DMODEL, DINNER);

    (void)in_sizes; (void)n_in; (void)out_size; (void)ws_size;
}
